// iRMB_42073499631852
// MI455X (gfx1250) — matrix operand order repaired and checked by reading, not compiled, not run on hardware
//
#include <hip/hip_runtime.h>

// ---------------------------------------------------------------------------
// Fused iRMB block for MI455X (gfx1250, wave32, WMMA bf16 16x16x32).
// One workgroup (8 wave32) per 7x7 grid-window; all intermediates in LDS.
// sched_group_barrier packs each tile as (16x ds_read)(8x wmma) so the XDL
// pipe sees back-to-back matrix ops instead of load-wait-wmma lock-step.
// ---------------------------------------------------------------------------

#define DEVINL __device__ __forceinline__

#if defined(__has_builtin)
#if __has_builtin(__builtin_amdgcn_sched_group_barrier)
#define SCHED_GROUP(mask, size) __builtin_amdgcn_sched_group_barrier((mask), (size), 0)
#endif
#endif
#ifndef SCHED_GROUP
#define SCHED_GROUP(mask, size)
#endif
#define SG_WMMA    0x008   // matrix ops
#define SG_DSREAD  0x100   // LDS reads

typedef __bf16 bf16;
typedef __attribute__((ext_vector_type(8)))  bf16  bf16x8;
typedef __attribute__((ext_vector_type(16))) bf16  bf16x16;
typedef __attribute__((ext_vector_type(8)))  float f32x8;

constexpr int CIN  = 256;   // channels in / CMID
constexpr int HWP  = 49;    // 7*7 positions per window
constexpr int MP   = 64;    // padded row count (4 x 16 tiles)
constexpr int LDX  = 264;   // LDS stride for 256-wide buffers (bf16 elems)
constexpr int LDQ  = 72;    // LDS stride for 64-wide per-head buffers
constexpr int NQK  = 4 * MP * LDQ;   // 18432 elems
constexpr int NVB  = MP * LDX;       // 16896 elems
constexpr size_t SMEM_BYTES = (size_t)(3 * NQK + NVB) * sizeof(bf16); // 144384 B

DEVINL bf16x16 cat8(bf16x8 lo, bf16x8 hi) {
    return __builtin_shufflevector(lo, hi, 0,1,2,3,4,5,6,7,8,9,10,11,12,13,14,15);
}

DEVINL f32x8 wmma_bf16(bf16x16 a, bf16x16 b, f32x8 c) {
    // D = A(16x32) * B(32x16) + C(16x16), f32 accumulate
    return __builtin_amdgcn_wmma_f32_16x16x32_bf16(false, a, false, b, (short)0, c, false, false);
}

// 16-bit A fragment, 16x32 tile from row-major LDS [row][k], stride ld.
DEVINL bf16x16 load_a_frag(const bf16* base, int m_base, int k_base, int ld, int lane) {
    const int m  = m_base + (lane & 15);
    const int k0 = k_base + ((lane >> 4) << 4);
    const bf16* p = base + m * ld + k0;
    bf16x8 lo = *(const bf16x8*)(p);
    bf16x8 hi = *(const bf16x8*)(p + 8);
    return cat8(lo, hi);
}

// 16-bit B fragment, 32x16 tile; source stored [n][k] (contiguous along K).
DEVINL bf16x16 load_b_frag_rowK(const bf16* base, int n_base, int k_base, int ld, int lane) {
    const int n  = n_base + (lane & 15);
    const int k0 = k_base + ((lane >> 4) << 4);
    const bf16* p = base + n * ld + k0;
    bf16x8 lo = *(const bf16x8*)(p);
    bf16x8 hi = *(const bf16x8*)(p + 8);
    return cat8(lo, hi);
}

// B fragment from fp32 global weights, row-major [o][k] (16 contiguous floats
// per lane -> L2-friendly), converted to bf16 on the fly.
DEVINL bf16x16 load_b_frag_gw(const float* __restrict__ W, int o_base, int k_base,
                              int ldk, int lane) {
    const float* p = W + (size_t)(o_base + (lane & 15)) * ldk + k_base + ((lane >> 4) << 4);
    bf16x16 r;
    #pragma unroll
    for (int j = 0; j < 16; ++j) r[j] = (bf16)p[j];
    return r;
}

// Store f32 accumulator tile to LDS as bf16.  VGPR g: lanes0-15 M=g, N=lane;
// lanes16-31 M=g+8, N=lane-16.
DEVINL void store_acc_bf16(bf16* base, int m_base, int n_base, int ld, int lane,
                           const f32x8& acc, float scale) {
    const int n  = n_base + (lane & 15);
    const int mb = m_base + ((lane >> 4) << 3);
    #pragma unroll
    for (int g = 0; g < 8; ++g)
        base[(mb + g) * ld + n] = (bf16)(acc[g] * scale);
}

__global__ __launch_bounds__(256)
void irmb_fused(const float* __restrict__ x,
                const float* __restrict__ bn_gamma, const float* __restrict__ bn_beta,
                const float* __restrict__ bn_mean,  const float* __restrict__ bn_var,
                const float* __restrict__ w_qk,     const float* __restrict__ w_v,
                const float* __restrict__ w_conv3,  const float* __restrict__ w_conv5,
                const float* __restrict__ w_conv7,
                const float* __restrict__ attn_w,   const float* __restrict__ c3_w,
                const float* __restrict__ c5_w,     const float* __restrict__ c7_w,
                const float* __restrict__ w_proj,
                float* __restrict__ out)
{
    extern __shared__ __align__(16) char smem_raw[];
    bf16* bufA = (bf16*)smem_raw;      // X (64x256, ld 264) then S probs (4x64x64, ld 72)
    bf16* bufQ = bufA + NQK;           // Q (4x64x64, ld 72) then fused act (64x256, ld 264)
    bf16* bufK = bufQ + NQK;           // K (4x64x64, ld 72) then V^T (256x64, ld 72)
    bf16* bufV = bufK + NQK;           // V (64x256, ld 264)

    const int tid  = threadIdx.x;
    const int lane = tid & 31;
    // wave id is wave-uniform: force it scalar so tile-selection branches stay SALU
    const int wid  = __builtin_amdgcn_readfirstlane(tid >> 5);

    const int bw = blockIdx.x;         // 2048 windows
    const int bi = bw >> 6;            // batch
    const int n1 = (bw >> 3) & 7;      // window phase y
    const int n2 = bw & 7;             // window phase x

    // Warm L2 for the weight matrices (global_prefetch_b8 path).
    __builtin_prefetch(w_qk  + tid * 512, 0, 0);
    __builtin_prefetch(w_v   + tid * 256, 0, 0);
    __builtin_prefetch(w_proj+ tid * 256, 0, 0);

    // ---- Phase 0: gather window + BatchNorm -> bf16 X in LDS (rows>=49 zero)
    {
        const int c = tid;
        const float g  = bn_gamma[c] * rsqrtf(bn_var[c] + 1e-5f);
        const float mu = bn_mean[c];
        const float be = bn_beta[c];
        const float* xb = x + ((size_t)bi * CIN + c) * 3136;
        #pragma unroll 4
        for (int p = 0; p < MP; ++p) {
            float v = 0.f;
            if (p < HWP) {
                const int h1 = p / 7, w1 = p % 7;
                v = (xb[(h1 * 8 + n1) * 56 + (w1 * 8 + n2)] - mu) * g + be;
            }
            bufA[p * LDX + c] = (bf16)v;
        }
    }
    __syncthreads();

    // ---- Phase 1: GEMM1  X(64x256) x {W_qk^T (256x512), W_v^T (256x256)}
    // 48 n-tiles of 16 outputs; wave w owns tiles {w, w+8, ..., w+40}.
    #pragma unroll 1
    for (int i = 0; i < 6; ++i) {
        const int nt = wid + i * 8;                 // 0..47 (wave-uniform)
        const bool is_v = (nt >= 32);
        const float* W = is_v ? w_v : w_qk;
        const int o_base = is_v ? (nt - 32) * 16 : nt * 16;

        bf16x16 bfrag[8];
        #pragma unroll
        for (int kc = 0; kc < 8; ++kc)
            bfrag[kc] = load_b_frag_gw(W, o_base, kc * 32, CIN, lane);

        #pragma unroll 1
        for (int mt = 0; mt < 4; ++mt) {
            // preload all A fragments, then issue 8 WMMAs back-to-back
            bf16x16 afrag[8];
            #pragma unroll
            for (int kc = 0; kc < 8; ++kc)
                afrag[kc] = load_a_frag(bufA, mt * 16, kc * 32, LDX, lane);
            f32x8 acc = {};
            #pragma unroll
            for (int kc = 0; kc < 8; ++kc)
                acc = wmma_bf16(afrag[kc], bfrag[kc], acc);
            SCHED_GROUP(SG_DSREAD, 16);   // 8 frags x 2 b128 reads
            SCHED_GROUP(SG_WMMA,    8);   // then 8 dense WMMAs

            // scatter: qk -> per-head Q (scaled 1/sqrt(64)) / K ; v -> V
            const int n  = lane & 15;
            const int mb = mt * 16 + ((lane >> 4) << 3);
            if (is_v) {
                #pragma unroll
                for (int g = 0; g < 8; ++g)
                    bufV[(mb + g) * LDX + o_base + n] = (bf16)acc[g];
            } else {
                const int oo = nt * 16 + n;         // 0..511
                const int h  = (oo >> 6) & 3;
                const int d  = oo & 63;
                bf16* dst = (nt < 16) ? bufQ : bufK;          // wave-uniform select
                const float sc = (nt < 16) ? 0.125f : 1.0f;   // 1/sqrt(DH) on Q
                #pragma unroll
                for (int g = 0; g < 8; ++g)
                    dst[(h * MP + mb + g) * LDQ + d] = (bf16)(acc[g] * sc);
            }
        }
    }
    __syncthreads();

    // ---- Phase 2: GEMM2  S[h] = Q[h] (64x64) x K[h]^T (64x64); K=64 -> 2 chunks
    bf16* bufS = bufA;   // X is dead, reuse for scores/probs
    #pragma unroll 1
    for (int i = 0; i < 8; ++i) {
        const int t  = wid * 8 + i;                 // 0..63
        const int h  = t >> 4;
        const int mt = (t >> 2) & 3;
        const int nt = t & 3;
        const bf16* Qh = bufQ + h * MP * LDQ;
        const bf16* Kh = bufK + h * MP * LDQ;
        bf16x16 a0 = load_a_frag(Qh, mt * 16, 0,  LDQ, lane);
        bf16x16 a1 = load_a_frag(Qh, mt * 16, 32, LDQ, lane);
        bf16x16 b0 = load_b_frag_rowK(Kh, nt * 16, 0,  LDQ, lane);
        bf16x16 b1 = load_b_frag_rowK(Kh, nt * 16, 32, LDQ, lane);
        f32x8 acc = {};
        acc = wmma_bf16(a0, b0, acc);
        acc = wmma_bf16(a1, b1, acc);
        SCHED_GROUP(SG_DSREAD, 8);
        SCHED_GROUP(SG_WMMA,   2);
        store_acc_bf16(bufS, mt * 16, nt * 16, LDQ, lane, acc, 1.0f);
    }
    __syncthreads();

    // ---- Phase 3: softmax (row per thread) + V transpose into bufK (K dead)
    {
        // transpose V (64x256, ld LDX) -> V^T (256x64, ld LDQ) for b128 B-frags
        const int c = tid;
        bf16* vt = bufK + c * LDQ;
        #pragma unroll 8
        for (int p = 0; p < MP; ++p) vt[p] = bufV[p * LDX + c];

        const int h = tid >> 6, r = tid & 63;
        bf16* row = bufS + (h * MP + r) * LDQ;
        float mx = -1e30f;
        #pragma unroll 7
        for (int j = 0; j < HWP; ++j) mx = fmaxf(mx, (float)row[j]);
        float sum = 0.f;
        #pragma unroll 7
        for (int j = 0; j < HWP; ++j) {
            float e = __expf((float)row[j] - mx);
            sum += e;
            row[j] = (bf16)e;
        }
        const float inv = 1.0f / sum;
        #pragma unroll 7
        for (int j = 0; j < HWP; ++j) row[j] = (bf16)((float)row[j] * inv);
        #pragma unroll
        for (int j = HWP; j < MP; ++j) row[j] = (bf16)0.f;   // mask pad columns
    }
    __syncthreads();

    // ---- Phase 4: GEMM3  x_spa[h] = S[h] (64x64) x V[:, h*64:(h+1)*64]
    bf16* bufVT = bufK;  // V^T (channel-major, contiguous along positions)
    bf16* bufF  = bufQ;  // Q is dead, reuse for attention-out / fused activation
    #pragma unroll 1
    for (int i = 0; i < 8; ++i) {
        const int t  = wid * 8 + i;
        const int h  = t >> 4;
        const int mt = (t >> 2) & 3;
        const int nt = t & 3;
        const bf16* Sh = bufS + h * MP * LDQ;
        bf16x16 a0 = load_a_frag(Sh, mt * 16, 0,  LDQ, lane);
        bf16x16 a1 = load_a_frag(Sh, mt * 16, 32, LDQ, lane);
        // B[k=pos][n=chan] <- V^T rows are channels, contiguous along pos
        bf16x16 b0 = load_b_frag_rowK(bufVT, h * 64 + nt * 16, 0,  LDQ, lane);
        bf16x16 b1 = load_b_frag_rowK(bufVT, h * 64 + nt * 16, 32, LDQ, lane);
        f32x8 acc = {};
        acc = wmma_bf16(a0, b0, acc);
        acc = wmma_bf16(a1, b1, acc);
        SCHED_GROUP(SG_DSREAD, 8);
        SCHED_GROUP(SG_WMMA,   2);
        store_acc_bf16(bufF, mt * 16, h * 64 + nt * 16, LDX, lane, acc, 1.0f);
    }
    __syncthreads();

    // ---- Phase 5: depthwise 3/5/7 convs on V + weighted fusion (+V residual)
    // one channel per thread; window held in registers, fully unrolled.
    {
        const int c = tid;
        const float aw = attn_w[c], f3 = c3_w[c], f5 = c5_w[c], f7 = c7_w[c];
        float k3[9], k5[25], k7[49], vv[49];
        #pragma unroll
        for (int j = 0; j < 9;  ++j) k3[j] = w_conv3[c * 9  + j];
        #pragma unroll
        for (int j = 0; j < 25; ++j) k5[j] = w_conv5[c * 25 + j];
        #pragma unroll
        for (int j = 0; j < 49; ++j) k7[j] = w_conv7[c * 49 + j];
        #pragma unroll
        for (int p = 0; p < HWP; ++p) vv[p] = (float)bufVT[c * LDQ + p];

        #pragma unroll
        for (int oy = 0; oy < 7; ++oy) {
            #pragma unroll
            for (int ox = 0; ox < 7; ++ox) {
                float s3 = 0.f, s5 = 0.f, s7 = 0.f;
                #pragma unroll
                for (int ky = 0; ky < 3; ++ky) {
                    const int iy = oy + ky - 1;
                    if (iy < 0 || iy >= 7) continue;
                    #pragma unroll
                    for (int kx = 0; kx < 3; ++kx) {
                        const int ix = ox + kx - 1;
                        if (ix < 0 || ix >= 7) continue;
                        s3 += vv[iy * 7 + ix] * k3[ky * 3 + kx];
                    }
                }
                #pragma unroll
                for (int ky = 0; ky < 5; ++ky) {
                    const int iy = oy + ky - 2;
                    if (iy < 0 || iy >= 7) continue;
                    #pragma unroll
                    for (int kx = 0; kx < 5; ++kx) {
                        const int ix = ox + kx - 2;
                        if (ix < 0 || ix >= 7) continue;
                        s5 += vv[iy * 7 + ix] * k5[ky * 5 + kx];
                    }
                }
                #pragma unroll
                for (int ky = 0; ky < 7; ++ky) {
                    const int iy = oy + ky - 3;
                    if (iy < 0 || iy >= 7) continue;
                    #pragma unroll
                    for (int kx = 0; kx < 7; ++kx) {
                        const int ix = ox + kx - 3;
                        if (ix < 0 || ix >= 7) continue;
                        s7 += vv[iy * 7 + ix] * k7[ky * 7 + kx];
                    }
                }
                const int p = oy * 7 + ox;
                const float spa = (float)bufF[p * LDX + c];
                bufF[p * LDX + c] =
                    (bf16)(spa * aw + s3 * f3 + s5 * f5 + s7 * f7 + vv[p]);
            }
        }
    }
    __syncthreads();

    // ---- Phase 6: GEMM4  out = fused (64x256) x W_proj^T (256x256) + shortcut
    #pragma unroll 1
    for (int i = 0; i < 8; ++i) {
        const int t  = wid * 8 + i;                 // 0..63
        const int mt = t >> 4;                      // 0..3
        const int nt = t & 15;                      // 0..15
        bf16x16 afrag[8];
        bf16x16 bfrag[8];
        #pragma unroll
        for (int kc = 0; kc < 8; ++kc) {
            afrag[kc] = load_a_frag(bufF, mt * 16, kc * 32, LDX, lane);
            bfrag[kc] = load_b_frag_gw(w_proj, nt * 16, kc * 32, CIN, lane);
        }
        f32x8 acc = {};
        #pragma unroll
        for (int kc = 0; kc < 8; ++kc)
            acc = wmma_bf16(afrag[kc], bfrag[kc], acc);
        SCHED_GROUP(SG_DSREAD, 16);
        SCHED_GROUP(SG_WMMA,    8);

        const int o  = nt * 16 + (lane & 15);
        const int mb = mt * 16 + ((lane >> 4) << 3);
        #pragma unroll
        for (int g = 0; g < 8; ++g) {
            const int p = mb + g;
            if (p < HWP) {
                const int h1 = p / 7, w1 = p % 7;
                const size_t idx = ((size_t)bi * CIN + o) * 3136
                                 + (size_t)(h1 * 8 + n1) * 56 + (w1 * 8 + n2);
                out[idx] = x[idx] + acc[g];
            }
        }
    }
}

extern "C" void kernel_launch(void* const* d_in, const int* in_sizes, int n_in,
                              void* d_out, int out_size, void* d_ws, size_t ws_size,
                              hipStream_t stream) {
    (void)in_sizes; (void)n_in; (void)d_ws; (void)ws_size; (void)out_size;
    const float* x        = (const float*)d_in[0];
    const float* bn_gamma = (const float*)d_in[1];
    const float* bn_beta  = (const float*)d_in[2];
    const float* bn_mean  = (const float*)d_in[3];
    const float* bn_var   = (const float*)d_in[4];
    const float* w_qk     = (const float*)d_in[5];
    const float* w_v      = (const float*)d_in[6];
    const float* w_conv3  = (const float*)d_in[7];
    const float* w_conv5  = (const float*)d_in[8];
    const float* w_conv7  = (const float*)d_in[9];
    const float* attn_w   = (const float*)d_in[10];
    const float* c3w      = (const float*)d_in[11];
    const float* c5w      = (const float*)d_in[12];
    const float* c7w      = (const float*)d_in[13];
    const float* w_proj   = (const float*)d_in[14];
    float* out            = (float*)d_out;

    hipFuncSetAttribute((const void*)irmb_fused,
                        hipFuncAttributeMaxDynamicSharedMemorySize,
                        (int)SMEM_BYTES);

    dim3 grid(2048), block(256);
    irmb_fused<<<grid, block, SMEM_BYTES, stream>>>(
        x, bn_gamma, bn_beta, bn_mean, bn_var, w_qk, w_v,
        w_conv3, w_conv5, w_conv7, attn_w, c3w, c5w, c7w, w_proj, out);
}
